// Proto_74217034874889
// MI455X (gfx1250) — compile-verified, matrix operand order checked by reading, not run on hardware
//
#include <hip/hip_runtime.h>
#include <hip/hip_bf16.h>
#include <cstdint>

// ---------- problem constants (reference: B=32768, D=1024, K=512, T=0.1) ----
#define B_ROWS 32768
#define D_IN   1024
#define K_OUT  512
#define INV_T  10.0f

// ---------- CDNA5 WMMA types ------------------------------------------------
typedef __attribute__((ext_vector_type(16))) __bf16 v16bf;
typedef __attribute__((ext_vector_type(8)))  float  v8f;

// fp32 -> bf16 round-to-nearest-even
__device__ __forceinline__ unsigned short f2bf(float f) {
  union { float f; unsigned u; } x; x.f = f;
  unsigned r = x.u + 0x7FFFu + ((x.u >> 16) & 1u);
  return (unsigned short)(r >> 16);
}

// Load one 16x32 bf16 WMMA fragment from an LDS row (row-major, K contiguous).
// Lane group kg = lane>>4 holds K = {kg*8..kg*8+7, 16+kg*8..16+kg*8+7}
// (ISA 7.12.2 "16-bit A-Matrix 16x32" layout); two ds_load_b128.
__device__ __forceinline__ v16bf frag_ld(const unsigned short* rowp, int kg) {
  union { v16bf v; uint4 u[2]; } f;
  f.u[0] = *(const uint4*)(rowp + kg * 8);
  f.u[1] = *(const uint4*)(rowp + 16 + kg * 8);
  return f.v;
}

// CDNA5 async global->LDS copy (16B per lane), tracked by ASYNCcnt.
// Flat address of a __shared__ object: addr[31:0] is the wave-relative LDS
// offset (aperture in the high bits), which is what VDST must carry.
__device__ __forceinline__ void async_copy16(void* ldsdst, const void* gsrc) {
  unsigned off = (unsigned)(unsigned long long)ldsdst;
  asm volatile("global_load_async_to_lds_b128 %0, %1, off"
               :: "v"(off), "v"(gsrc) : "memory");
}
__device__ __forceinline__ void async_wait0() {
  asm volatile("s_wait_asynccnt 0" ::: "memory");
}

#define BM 64    // block tile rows (block owns full 512-wide rows)
#define KC 32    // inner-dim chunk (bf16 WMMA K)
#define LD 40    // LDS row stride in bf16 elems (padded vs 32 -> no bank conflicts)

// ---------- init: zero loss scalar + S0[512] --------------------------------
__global__ void k_init(float* loss, float* S0) {
  if (threadIdx.x == 0) *loss = 0.f;
  S0[threadIdx.x] = 0.f;  // launched with 512 threads
}

// ---------- row L2-norm (optional) + fp32 -> bf16 ---------------------------
__global__ __launch_bounds__(256) void k_rownorm(const float* __restrict__ in,
                                                 unsigned short* __restrict__ out,
                                                 int cols, int donorm) {
  __shared__ float sm[256];
  const int row = blockIdx.x, tid = threadIdx.x;
  const float* rp = in + (size_t)row * cols;
  float ss = 0.f;
  for (int c = tid; c < cols; c += 256) { float v = rp[c]; ss += v * v; }
  sm[tid] = ss; __syncthreads();
  for (int st = 128; st > 0; st >>= 1) { if (tid < st) sm[tid] += sm[tid + st]; __syncthreads(); }
  const float sc = donorm ? rsqrtf(sm[0]) : 1.f;
  for (int c = tid; c < cols; c += 256) out[(size_t)row * cols + c] = f2bf(rp[c] * sc);
}

// ---------- GEMM1: Z = A(fp32 BxD) @ W(bf16 KxD)^T, fused row L2-norm, bf16 out
// 256 threads = 8 waves, 4M x 2N wave grid, each wave 16 x 256 -> 16 v8f accs.
__global__ __launch_bounds__(256) void k_gemm1(const float* __restrict__ A,
                                               const unsigned short* __restrict__ Wb,
                                               unsigned short* __restrict__ Zb) {
  __shared__ unsigned short As[BM * LD];      // 5 KB
  __shared__ unsigned short Bs[K_OUT * LD];   // 40 KB
  __shared__ float red[2 * BM];

  const int tid = threadIdx.x;
  const int lane = tid & 31, wave = tid >> 5;
  const int wm = wave >> 1, wn = wave & 1;
  const int half = lane >> 4, l16 = lane & 15, kg = half;
  const size_t rowBase = (size_t)blockIdx.x * BM;

  v8f acc[16];
  const v8f vz = {0.f, 0.f, 0.f, 0.f, 0.f, 0.f, 0.f, 0.f};
  #pragma unroll
  for (int n = 0; n < 16; ++n) acc[n] = vz;

  for (int kc = 0; kc < D_IN / KC; ++kc) {
    // B tile 512x32 bf16: async global->LDS (no VGPR round-trip)
    #pragma unroll
    for (int i = 0; i < 8; ++i) {
      int q = tid + i * 256, row = q >> 2, seg = q & 3;
      async_copy16(Bs + row * LD + seg * 8,
                   Wb + (size_t)row * D_IN + kc * KC + seg * 8);
    }
    // A tile 64x32 fp32 -> bf16: batch loads, then convert+store (one wait)
    float4 fa[2];
    #pragma unroll
    for (int i = 0; i < 2; ++i) {
      int q = tid + i * 256, row = q >> 3, seg = q & 7;
      fa[i] = *(const float4*)(A + (rowBase + row) * D_IN + kc * KC + seg * 4);
    }
    #pragma unroll
    for (int i = 0; i < 2; ++i) {
      int q = tid + i * 256, row = q >> 3, seg = q & 7;
      unsigned lo = (unsigned)f2bf(fa[i].x) | ((unsigned)f2bf(fa[i].y) << 16);
      unsigned hi = (unsigned)f2bf(fa[i].z) | ((unsigned)f2bf(fa[i].w) << 16);
      *(uint2*)(As + row * LD + seg * 4) = make_uint2(lo, hi);
    }
    async_wait0();
    __syncthreads();
    v16bf a = frag_ld(As + (wm * 16 + l16) * LD, kg);
    #pragma unroll
    for (int nt = 0; nt < 16; ++nt) {
      v16bf b = frag_ld(Bs + (wn * 256 + nt * 16 + l16) * LD, kg);
      acc[nt] = __builtin_amdgcn_wmma_f32_16x16x32_bf16(false, a, false, b,
                                                        (short)0, acc[nt], false, false);
    }
    __syncthreads();
  }

  // fused row L2-norm: block owns full 512-wide rows
  float ss[8];
  #pragma unroll
  for (int r = 0; r < 8; ++r) ss[r] = 0.f;
  #pragma unroll
  for (int nt = 0; nt < 16; ++nt)
    #pragma unroll
    for (int r = 0; r < 8; ++r) { float v = acc[nt][r]; ss[r] += v * v; }
  #pragma unroll
  for (int r = 0; r < 8; ++r)
    #pragma unroll
    for (int m = 1; m <= 8; m <<= 1) ss[r] += __shfl_xor(ss[r], m, 32);
  if (l16 == 0) {
    #pragma unroll
    for (int r = 0; r < 8; ++r) red[wn * BM + wm * 16 + half * 8 + r] = ss[r];
  }
  __syncthreads();
  float sc[8];
  #pragma unroll
  for (int r = 0; r < 8; ++r) {
    int rl = wm * 16 + half * 8 + r;
    sc[r] = rsqrtf(red[rl] + red[BM + rl]);
  }
  #pragma unroll
  for (int nt = 0; nt < 16; ++nt)
    #pragma unroll
    for (int r = 0; r < 8; ++r) {
      size_t grow = rowBase + wm * 16 + half * 8 + r;
      int col = wn * 256 + nt * 16 + l16;
      Zb[grow * K_OUT + col] = f2bf(acc[nt][r] * sc[r]);
    }
}

// ---------- GEMM2: scores = Z(bf16 BxK) @ Cn(bf16 KxK)^T --------------------
// mode 0 (student): fused log-softmax(scores/T) -> outp (fp32)
// mode 1 (teacher): Q0 = exp(scores/T) -> outp; column sums atomically into S0
__global__ __launch_bounds__(256) void k_gemm2(const unsigned short* __restrict__ Ab,
                                               const unsigned short* __restrict__ Cb,
                                               float* __restrict__ outp,
                                               float* __restrict__ S0, int mode) {
  __shared__ unsigned short As[BM * LD];
  __shared__ unsigned short Bs[K_OUT * LD];
  __shared__ float red[2 * BM];
  __shared__ float colss[K_OUT];

  const int tid = threadIdx.x;
  const int lane = tid & 31, wave = tid >> 5;
  const int wm = wave >> 1, wn = wave & 1;
  const int half = lane >> 4, l16 = lane & 15, kg = half;
  const size_t rowBase = (size_t)blockIdx.x * BM;

  v8f acc[16];
  const v8f vz = {0.f, 0.f, 0.f, 0.f, 0.f, 0.f, 0.f, 0.f};
  #pragma unroll
  for (int n = 0; n < 16; ++n) acc[n] = vz;

  for (int kc = 0; kc < K_OUT / KC; ++kc) {
    // A tile 64x32 bf16: 1 async b128 per thread
    {
      int row = tid >> 2, seg = tid & 3;
      async_copy16(As + row * LD + seg * 8,
                   Ab + (rowBase + row) * K_OUT + kc * KC + seg * 8);
    }
    // B tile 512x32 bf16 (Cn, L2-resident): 8 async b128 per thread
    #pragma unroll
    for (int i = 0; i < 8; ++i) {
      int q = tid + i * 256, row = q >> 2, seg = q & 3;
      async_copy16(Bs + row * LD + seg * 8,
                   Cb + (size_t)row * K_OUT + kc * KC + seg * 8);
    }
    async_wait0();
    __syncthreads();
    v16bf a = frag_ld(As + (wm * 16 + l16) * LD, kg);
    #pragma unroll
    for (int nt = 0; nt < 16; ++nt) {
      v16bf b = frag_ld(Bs + (wn * 256 + nt * 16 + l16) * LD, kg);
      acc[nt] = __builtin_amdgcn_wmma_f32_16x16x32_bf16(false, a, false, b,
                                                        (short)0, acc[nt], false, false);
    }
    __syncthreads();
  }

  if (mode == 0) {
    // fused log-softmax over the full 512-wide row
    float mx[8];
    #pragma unroll
    for (int r = 0; r < 8; ++r) mx[r] = -3.0e38f;
    #pragma unroll
    for (int nt = 0; nt < 16; ++nt)
      #pragma unroll
      for (int r = 0; r < 8; ++r) mx[r] = fmaxf(mx[r], acc[nt][r] * INV_T);
    #pragma unroll
    for (int r = 0; r < 8; ++r)
      #pragma unroll
      for (int m = 1; m <= 8; m <<= 1) mx[r] = fmaxf(mx[r], __shfl_xor(mx[r], m, 32));
    if (l16 == 0) {
      #pragma unroll
      for (int r = 0; r < 8; ++r) red[wn * BM + wm * 16 + half * 8 + r] = mx[r];
    }
    __syncthreads();
    float mrow[8];
    #pragma unroll
    for (int r = 0; r < 8; ++r) {
      int rl = wm * 16 + half * 8 + r;
      mrow[r] = fmaxf(red[rl], red[BM + rl]);
    }
    __syncthreads();
    float se[8];
    #pragma unroll
    for (int r = 0; r < 8; ++r) se[r] = 0.f;
    #pragma unroll
    for (int nt = 0; nt < 16; ++nt)
      #pragma unroll
      for (int r = 0; r < 8; ++r) se[r] += __expf(acc[nt][r] * INV_T - mrow[r]);
    #pragma unroll
    for (int r = 0; r < 8; ++r)
      #pragma unroll
      for (int m = 1; m <= 8; m <<= 1) se[r] += __shfl_xor(se[r], m, 32);
    if (l16 == 0) {
      #pragma unroll
      for (int r = 0; r < 8; ++r) red[wn * BM + wm * 16 + half * 8 + r] = se[r];
    }
    __syncthreads();
    #pragma unroll
    for (int r = 0; r < 8; ++r) {
      int rl = wm * 16 + half * 8 + r;
      float lz = __logf(red[rl] + red[BM + rl]);
      size_t grow = rowBase + rl;
      #pragma unroll
      for (int nt = 0; nt < 16; ++nt) {
        int col = wn * 256 + nt * 16 + l16;
        outp[grow * K_OUT + col] = acc[nt][r] * INV_T - mrow[r] - lz;
      }
    }
  } else {
    // teacher: Q0 = exp(scores/T) (global max/sum cancel in Sinkhorn; exp<=e^10 safe)
    // + fused column sums -> S0 (first Sinkhorn row-sum pass, V0 == 1)
    float cp[16];
    #pragma unroll
    for (int nt = 0; nt < 16; ++nt) cp[nt] = 0.f;
    #pragma unroll
    for (int nt = 0; nt < 16; ++nt)
      #pragma unroll
      for (int r = 0; r < 8; ++r) {
        float q = __expf(acc[nt][r] * INV_T);
        size_t grow = rowBase + wm * 16 + half * 8 + r;
        int col = wn * 256 + nt * 16 + l16;
        outp[grow * K_OUT + col] = q;
        cp[nt] += q;
      }
    #pragma unroll
    for (int nt = 0; nt < 16; ++nt) cp[nt] += __shfl_xor(cp[nt], 16, 32);  // both halves
    colss[tid] = 0.f; colss[tid + 256] = 0.f;
    __syncthreads();
    if (half == 0) {
      #pragma unroll
      for (int nt = 0; nt < 16; ++nt)
        atomicAdd(&colss[wn * 256 + nt * 16 + l16], cp[nt]);
    }
    __syncthreads();
    atomicAdd(&S0[tid], colss[tid]);
    atomicAdd(&S0[tid + 256], colss[tid + 256]);
  }
}

// ---------- Sinkhorn vector passes ------------------------------------------
// S[k] += sum_b Q0[b,k] * V[b]   (column pass; each thread owns 2 columns)
__global__ __launch_bounds__(256) void k_colpass(const float* __restrict__ Q0,
                                                 const float* __restrict__ V,
                                                 float* __restrict__ S) {
  const int c0 = threadIdx.x * 2;
  const size_t r0 = (size_t)blockIdx.x * 128;
  float a0 = 0.f, a1 = 0.f;
  #pragma unroll 4
  for (int i = 0; i < 128; ++i) {
    size_t b = r0 + i;
    float v = V[b];
    float2 q = *(const float2*)(Q0 + b * K_OUT + c0);
    a0 += q.x * v; a1 += q.y * v;
  }
  atomicAdd(&S[c0], a0);
  atomicAdd(&S[c0 + 1], a1);
}

// V[b] = c / sum_k Q0[b,k] * U[k]   (row pass; one wave per row)
__global__ __launch_bounds__(256) void k_rowpass(const float* __restrict__ Q0,
                                                 const float* __restrict__ U,
                                                 float* __restrict__ V, float cval) {
  const int lane = threadIdx.x & 31, wave = threadIdx.x >> 5;
  const size_t b = (size_t)blockIdx.x * 8 + wave;
  float s = 0.f;
  #pragma unroll
  for (int i = 0; i < 16; ++i) { int k = i * 32 + lane; s += Q0[b * K_OUT + k] * U[k]; }
  #pragma unroll
  for (int m = 1; m <= 16; m <<= 1) s += __shfl_xor(s, m, 32);
  if (lane == 0) V[b] = cval / s;
}

// U[k] = r / S[k]; zero the next S buffer
__global__ void k_uk(const float* S, float* U, float rval, float* Sz) {
  const int k = threadIdx.x;  // 512 threads
  U[k] = rval / S[k];
  if (Sz) Sz[k] = 0.f;
}

// final: Qf[b,k] = Q0*U / (Q0^T U)[b];  loss -= mean_b sum_k Qf * logp
__global__ __launch_bounds__(256) void k_loss(const float* __restrict__ Q0,
                                              const float* __restrict__ U,
                                              const float* __restrict__ LP,
                                              float* loss) {
  const int lane = threadIdx.x & 31, wave = threadIdx.x >> 5;
  const size_t b = (size_t)blockIdx.x * 8 + wave;
  float d = 0.f, e = 0.f;
  #pragma unroll
  for (int i = 0; i < 16; ++i) {
    int k = i * 32 + lane;
    float qu = Q0[b * K_OUT + k] * U[k];
    d += qu; e += qu * LP[b * K_OUT + k];
  }
  #pragma unroll
  for (int m = 1; m <= 16; m <<= 1) { d += __shfl_xor(d, m, 32); e += __shfl_xor(e, m, 32); }
  __shared__ float part[8];
  if (lane == 0) part[wave] = e / d;
  __syncthreads();
  if (threadIdx.x == 0) {
    float acc = 0.f;
    for (int w = 0; w < 8; ++w) acc += part[w];
    atomicAdd(loss, -acc / (float)B_ROWS);
  }
}

// ---------- host-side launch -------------------------------------------------
extern "C" void kernel_launch(void* const* d_in, const int* in_sizes, int n_in,
                              void* d_out, int out_size, void* d_ws, size_t ws_size,
                              hipStream_t stream) {
  (void)in_sizes; (void)n_in; (void)out_size; (void)ws_size;
  const float* s  = (const float*)d_in[0];   // [B, D]
  const float* t  = (const float*)d_in[1];   // [B, D]
  const float* Wn = (const float*)d_in[2];   // [K, D]
  const float* Wp = (const float*)d_in[3];   // [K, K]
  float* loss = (float*)d_out;

  char* ws = (char*)d_ws;
  size_t off = 0;
  auto alloc = [&](size_t bytes) -> void* {
    void* p = ws + off;
    off = (off + bytes + 255) & ~(size_t)255;
    return p;
  };
  unsigned short* Wnb = (unsigned short*)alloc((size_t)K_OUT * D_IN * 2);   // 1 MB
  unsigned short* Cnb = (unsigned short*)alloc((size_t)K_OUT * K_OUT * 2);  // 0.5 MB
  unsigned short* Zs  = (unsigned short*)alloc((size_t)B_ROWS * K_OUT * 2); // 32 MB
  unsigned short* Zt  = (unsigned short*)alloc((size_t)B_ROWS * K_OUT * 2); // 32 MB
  float* LP = (float*)alloc((size_t)B_ROWS * K_OUT * 4);                    // 64 MB
  float* Q0 = (float*)alloc((size_t)B_ROWS * K_OUT * 4);                    // 64 MB
  float* S0 = (float*)alloc(K_OUT * 4);
  float* S1 = (float*)alloc(K_OUT * 4);
  float* S2 = (float*)alloc(K_OUT * 4);
  float* U  = (float*)alloc(K_OUT * 4);
  float* V  = (float*)alloc((size_t)B_ROWS * 4);

  const float rv = 1.f / (float)K_OUT, cv = 1.f / (float)B_ROWS;
  const int GB = B_ROWS / BM;  // 512 GEMM blocks

  k_init<<<1, 512, 0, stream>>>(loss, S0);
  k_rownorm<<<K_OUT, 256, 0, stream>>>(Wn, Wnb, D_IN, 0);   // W_net -> bf16
  k_rownorm<<<K_OUT, 256, 0, stream>>>(Wp, Cnb, K_OUT, 1);  // Cn = l2norm(W_proto)

  k_gemm1<<<GB, 256, 0, stream>>>(s, Wnb, Zs);              // zs (normalized, bf16)
  k_gemm1<<<GB, 256, 0, stream>>>(t, Wnb, Zt);              // zt

  k_gemm2<<<GB, 256, 0, stream>>>(Zs, Cnb, LP, nullptr, 0); // log-softmax
  k_gemm2<<<GB, 256, 0, stream>>>(Zt, Cnb, Q0, S0, 1);      // Q0 + col sums

  // Sinkhorn as U/V recurrences: U1=r/S0; V=c/(Q0 U); S=Q0^T V; ... (3 iters)
  k_uk<<<1, 512, 0, stream>>>(S0, U, rv, S1);               // U1, zero S1
  k_rowpass<<<B_ROWS / 8, 256, 0, stream>>>(Q0, U, V, cv);  // V1
  k_colpass<<<B_ROWS / 128, 256, 0, stream>>>(Q0, V, S1);   // S1
  k_uk<<<1, 512, 0, stream>>>(S1, U, rv, S2);               // U2, zero S2
  k_rowpass<<<B_ROWS / 8, 256, 0, stream>>>(Q0, U, V, cv);  // V2
  k_colpass<<<B_ROWS / 128, 256, 0, stream>>>(Q0, V, S2);   // S2
  k_uk<<<1, 512, 0, stream>>>(S2, U, rv, nullptr);          // U3

  k_loss<<<B_ROWS / 8, 256, 0, stream>>>(Q0, U, LP, loss);  // final normalize + loss
}
